// OneRec_65266323030562
// MI455X (gfx1250) — compile-verified, matrix operand order checked by reading, not run on hardware
//
#include <hip/hip_runtime.h>
#include <hip/hip_bf16.h>
#include <cstdint>

// ---------------- model constants (match reference) ----------------
#define DMODEL 1024
#define NHEAD  16
#define NLAYER 2
#define NEXP   8
#define HIDF   4096
#define KCB    8192
#define LLEV   3
#define BB     8
#define TT     128
#define TEE    256
#define NTOK   (BB*TT)      // 1024 decoder tokens
#define NKV    (BB*TEE)     // 2048 encoder tokens
#define DH     64           // head dim

typedef __attribute__((ext_vector_type(16))) __bf16 v16bf;
typedef __attribute__((ext_vector_type(4)))  __bf16 bf16x4;
typedef __attribute__((ext_vector_type(8)))  float  v8f;
typedef __attribute__((ext_vector_type(4)))  float  f32x4;
typedef __attribute__((ext_vector_type(4)))  unsigned int u32x4;
typedef __attribute__((ext_vector_type(8)))  int i32x8;
typedef __attribute__((ext_vector_type(4)))  int i32x4;

union BF16Frag { v16bf v; unsigned int u[8]; };

// fp32 quad -> packed bf16 quad via vector convert (lets the backend select
// packed v_cvt_*bf16_f32 hardware conversion when available)
__device__ __forceinline__ uint2 packbf4(float4 x) {
  f32x4 v; v[0] = x.x; v[1] = x.y; v[2] = x.z; v[3] = x.w;
  const bf16x4 h = __builtin_convertvector(v, bf16x4);
  union { bf16x4 h; uint2 u2; } cv; cv.h = h;
  return cv.u2;
}

// generic pointer -> 32-bit LDS byte offset (for TDM descriptors)
typedef __attribute__((address_space(3))) const char lds_cptr_t;
__device__ __forceinline__ unsigned lds_off_u32(const void* p) {
  return (unsigned)(size_t)(lds_cptr_t*)p;
}

// ---------------------------------------------------------------------
// Tensor Data Mover: issue a 2-D tile load (rows x 32 fp32, row stride
// `strideElems` floats) from global -> LDS. D# per CDNA5 ISA ch.8:
//   group0: count=1 | lds_addr | global_addr(57b) | type=2
//   group1: data_size=4B, tensor dims (in-bounds by construction),
//           tile_dim0=32, tile_dim1=rows, dim0 stride
//   groups 2/3: zero (2-D tensor)
// ---------------------------------------------------------------------
__device__ __forceinline__ void tdm_load_tile(unsigned lds_addr, const float* gp,
                                              unsigned strideElems, unsigned rows) {
  const unsigned long long ga = (unsigned long long)(uintptr_t)gp;
  u32x4 g0;
  g0[0] = 1u;                                               // count=1
  g0[1] = lds_addr;                                         // LDS byte address
  g0[2] = (unsigned)(ga & 0xFFFFFFFFull);                   // global_addr lo
  g0[3] = (unsigned)((ga >> 32) & 0x01FFFFFFull) | (2u << 30); // addr hi | type=2
  const unsigned td0 = strideElems;                         // tensor_dim0 (elems)
  const unsigned td1 = 0x00100000u;                         // tensor_dim1 (large)
  const unsigned long long s0 = (unsigned long long)strideElems;
  i32x8 g1;
  g1[0] = (int)(2u << 16);                                  // data_size=4B, mask=0
  g1[1] = (int)((td0 & 0xFFFFu) << 16);                     // td0[15:0] @ [63:48]
  g1[2] = (int)(((td0 >> 16) & 0xFFFFu) | ((td1 & 0xFFFFu) << 16));
  g1[3] = (int)(((td1 >> 16) & 0xFFFFu) | (32u << 16));     // tile_dim0=32 @[127:112]
  g1[4] = (int)(rows & 0xFFFFu);                            // tile_dim1=rows, tile_dim2=0
  g1[5] = (int)(s0 & 0xFFFFFFFFull);                        // dim0 stride lo
  g1[6] = (int)((s0 >> 32) & 0xFFFFull);                    // stride hi | dim1 stride lo=0
  g1[7] = 0;
  const i32x4 z4 = {0, 0, 0, 0};
#if __has_include(<hip/amd_detail/amd_gfx1250_TDM.h>)
  const i32x8 z8 = {0, 0, 0, 0, 0, 0, 0, 0};
  __builtin_amdgcn_tensor_load_to_lds(g0, g1, z4, z4, z8, 0);
#else
  __builtin_amdgcn_tensor_load_to_lds(g0, g1, z4, z4, 0);
#endif
}

// load one 16x32 bf16 fragment from a packed-pair LDS row (16 uints/row):
// lanes 0-15 hold K {0..7,16..23}, lanes 16-31 hold K {8..15,24..31}
// -> two contiguous uint4 loads at pair offsets kh*4 and 8+kh*4
__device__ __forceinline__ void load_frag(const unsigned int* row16, int kh, BF16Frag& f) {
  const uint4 u0 = *(const uint4*)(row16 + kh * 4);
  const uint4 u1 = *(const uint4*)(row16 + 8 + kh * 4);
  f.u[0] = u0.x; f.u[1] = u0.y; f.u[2] = u0.z; f.u[3] = u0.w;
  f.u[4] = u1.x; f.u[5] = u1.y; f.u[6] = u1.z; f.u[7] = u1.w;
}

// =====================================================================
// TDM-fed bf16 WMMA GEMM:  C[N x M] = A[N x K] * W[M x K]^T (fp32 in/out)
// 256 threads = 8 waves. Workgroup tile 128(rows) x 64(cols).
// Wave (wr,wc): wr in 0..3 -> 32-row strip, wc in 0..1 -> 32 cols
//   -> 4 WMMA accumulators (2 A-frags x 2 B-frags) per wave per K-step.
// Pipeline per K-step:
//   wave0: TDM-issue next fp32 tiles (double buffered) + s_wait_tensorcnt
//   barrier A | cooperative fp32->packed-bf16 convert (once per element)
//   barrier B | packed ds_load fragments + 4 WMMAs
// LDS: 2x(16+8) KB fp32 staging + (8+4) KB bf16 tile = 60 KB.
// MODE: 0 store | 1 C+=acc | 2 C+=gate[row]*acc | 3 heads [L,B,T,K] layout
// Requires N % 128 == 0, M % 64 == 0, K % 32 == 0 (true for all calls).
// =====================================================================
template<int MODE>
__global__ __launch_bounds__(256)
void gemm_bf16_tn(const float* __restrict__ A, const float* __restrict__ W,
                  float* __restrict__ C, int N, int M, int K,
                  const float* __restrict__ gate, int gstride) {
  __shared__ float lAf[2][128 * 32];        // TDM fp32 staging, 2 x 16 KB
  __shared__ float lBf[2][64 * 32];         // TDM fp32 staging, 2 x  8 KB
  __shared__ unsigned int lAh[128 * 16];    // packed bf16 tile,     8 KB
  __shared__ unsigned int lBh[64 * 16];     // packed bf16 tile,     4 KB

  const int tid  = threadIdx.x;
  const int row0 = blockIdx.y * 128;
  const int col0 = blockIdx.x * 64;
  const int wave = tid >> 5, lane = tid & 31;
  const int wr = wave & 3, wc = wave >> 2;
  const int lm = lane & 15, kh = lane >> 4;

  v8f acc00 = {}; v8f acc01 = {}; v8f acc10 = {}; v8f acc11 = {};

  const int nIter = K >> 5;
  if (wave == 0) {   // prologue: fill fp32 buffer 0 via DMA
    tdm_load_tile(lds_off_u32(&lAf[0][0]), A + (size_t)row0 * K, (unsigned)K, 128u);
    tdm_load_tile(lds_off_u32(&lBf[0][0]), W + (size_t)col0 * K, (unsigned)K, 64u);
  }

  for (int it = 0; it < nIter; ++it) {
    const int cur = it & 1;
    if (wave == 0) {
      if (it + 1 < nIter) {   // prefetch next K-tile into the other fp32 buffer
        const int kn = (it + 1) * 32;
        tdm_load_tile(lds_off_u32(&lAf[1 - cur][0]), A + (size_t)row0 * K + kn,
                      (unsigned)K, 128u);
        tdm_load_tile(lds_off_u32(&lBf[1 - cur][0]), W + (size_t)col0 * K + kn,
                      (unsigned)K, 64u);
        __builtin_amdgcn_s_wait_tensorcnt(2);   // current buffer's 2 DMAs done
      } else {
        __builtin_amdgcn_s_wait_tensorcnt(0);
      }
    }
    __syncthreads();   // barrier A: fp32[cur] published; prior frag reads retired

    // cooperative fp32 -> packed bf16 (each tile element converted exactly once)
    {
      const float* sa = &lAf[cur][tid * 16];
      unsigned int* da = &lAh[tid * 8];
#pragma unroll
      for (int i = 0; i < 4; ++i) {
        const uint2 p = packbf4(*(const float4*)(sa + i * 4));
        da[i * 2 + 0] = p.x;
        da[i * 2 + 1] = p.y;
      }
      const float* sb = &lBf[cur][tid * 8];
      unsigned int* db = &lBh[tid * 4];
#pragma unroll
      for (int i = 0; i < 2; ++i) {
        const uint2 p = packbf4(*(const float4*)(sb + i * 4));
        db[i * 2 + 0] = p.x;
        db[i * 2 + 1] = p.y;
      }
    }
    __syncthreads();   // barrier B: bf16 tile published

    BF16Frag a0, a1, b0, b1;
    load_frag(&lAh[(wr * 32 +      lm) * 16], kh, a0);
    load_frag(&lAh[(wr * 32 + 16 + lm) * 16], kh, a1);
    load_frag(&lBh[(wc * 32 +      lm) * 16], kh, b0);
    load_frag(&lBh[(wc * 32 + 16 + lm) * 16], kh, b1);

    acc00 = __builtin_amdgcn_wmma_f32_16x16x32_bf16(false, a0.v, false, b0.v,
                                                    (short)0, acc00, false, false);
    acc01 = __builtin_amdgcn_wmma_f32_16x16x32_bf16(false, a0.v, false, b1.v,
                                                    (short)0, acc01, false, false);
    acc10 = __builtin_amdgcn_wmma_f32_16x16x32_bf16(false, a1.v, false, b0.v,
                                                    (short)0, acc10, false, false);
    acc11 = __builtin_amdgcn_wmma_f32_16x16x32_bf16(false, a1.v, false, b1.v,
                                                    (short)0, acc11, false, false);
  }

  // C/D layout: lanes 0-15: row=r, col=lane; lanes 16-31: row=r+8, col=lane-16
  const int c0i = col0 + wc * 32 + lm;
  const int c1i = c0i + 16;
#pragma unroll
  for (int i = 0; i < 2; ++i) {
    const int rbase = row0 + wr * 32 + i * 16 + kh * 8;
    const v8f* va = (i == 0) ? &acc00 : &acc10;
    const v8f* vb = (i == 0) ? &acc01 : &acc11;
#pragma unroll
    for (int r = 0; r < 8; ++r) {
      const int rr = rbase + r;
      const float v0 = (*va)[r], v1 = (*vb)[r];
      if (MODE == 0) {
        C[(size_t)rr * M + c0i] = v0;
        C[(size_t)rr * M + c1i] = v1;
      } else if (MODE == 1) {
        C[(size_t)rr * M + c0i] += v0;
        C[(size_t)rr * M + c1i] += v1;
      } else if (MODE == 2) {
        const float g = gate[(size_t)rr * gstride];
        C[(size_t)rr * M + c0i] += g * v0;
        C[(size_t)rr * M + c1i] += g * v1;
      } else {
        const int l0 = c0i >> 13, k0c = c0i & (KCB - 1);
        const int l1 = c1i >> 13, k1c = c1i & (KCB - 1);
        C[((size_t)l0 * N + rr) * KCB + k0c] = v0;
        C[((size_t)l1 * N + rr) * KCB + k1c] = v1;
      }
    }
  }
}

// =====================================================================
// Embedding gather: x[n] = embed[ids[n]]
// =====================================================================
__global__ __launch_bounds__(256)
void embed_kernel(const int* __restrict__ ids, const float* __restrict__ emb,
                  float* __restrict__ X) {
  const int n = blockIdx.x, t = threadIdx.x;
  const int id = ids[n];
  const float4 v = *(const float4*)(emb + (size_t)id * DMODEL + t * 4);
  *(float4*)(X + (size_t)n * DMODEL + t * 4) = v;
}

// =====================================================================
// RMSNorm: one block per row, 256 threads x 4 elems
// =====================================================================
__global__ __launch_bounds__(256)
void rmsnorm_kernel(const float* __restrict__ X, const float* __restrict__ Wt,
                    float* __restrict__ Y) {
  __shared__ float red[256];
  const int row = blockIdx.x, t = threadIdx.x;
  const float4 xv = *(const float4*)(X + (size_t)row * DMODEL + t * 4);
  red[t] = xv.x * xv.x + xv.y * xv.y + xv.z * xv.z + xv.w * xv.w;
  __syncthreads();
  for (int s = 128; s > 0; s >>= 1) {
    if (t < s) red[t] += red[t + s];
    __syncthreads();
  }
  const float inv = rsqrtf(red[0] * (1.0f / DMODEL) + 1e-6f);
  const float4 wv = *(const float4*)(Wt + t * 4);
  float4 yv;
  yv.x = xv.x * inv * wv.x; yv.y = xv.y * inv * wv.y;
  yv.z = xv.z * inv * wv.z; yv.w = xv.w * inv * wv.w;
  *(float4*)(Y + (size_t)row * DMODEL + t * 4) = yv;
}

// =====================================================================
// Flash-style attention: one workgroup per (b,h), one thread per query row.
// K/V streamed through dynamic LDS in 128-row chunks (64 KB), online softmax.
// =====================================================================
__global__ void attn_kernel(const float* __restrict__ Qb, const float* __restrict__ Kb,
                            const float* __restrict__ Vb, float* __restrict__ Ob,
                            int Tq, int Tk, int causal) {
  extern __shared__ float smem[];              // kbuf[128*64] | vbuf[128*64]
  float* kbuf = smem;
  float* vbuf = smem + 128 * DH;
  const int bh = blockIdx.x;
  const int b = bh / NHEAD, h = bh % NHEAD;
  const int t = threadIdx.x;                   // query row, blockDim.x == Tq

  float q[DH], o[DH];
  const float* qp = Qb + ((size_t)(b * Tq + t)) * DMODEL + h * DH;
#pragma unroll
  for (int i = 0; i < DH; ++i) { q[i] = qp[i]; o[i] = 0.0f; }

  const float scale = 0.125f;                  // 1/sqrt(64)
  float m = -1e30f, ssum = 0.0f;

  for (int c0 = 0; c0 < Tk; c0 += 128) {
    const int cn = (Tk - c0 < 128) ? (Tk - c0) : 128;
    __syncthreads();
    for (int idx = t; idx < cn * 16; idx += Tq) {
      const int j = idx >> 4, c4 = (idx & 15) * 4;
      const size_t src = ((size_t)(b * Tk + c0 + j)) * DMODEL + h * DH + c4;
      *(float4*)(kbuf + j * DH + c4) = *(const float4*)(Kb + src);
      *(float4*)(vbuf + j * DH + c4) = *(const float4*)(Vb + src);
    }
    __syncthreads();

    int jlim = causal ? (t + 1 - c0) : cn;
    if (jlim > cn) jlim = cn;
    for (int j = 0; j < jlim; ++j) {
      const float* kr = kbuf + j * DH;
      float s = 0.0f;
#pragma unroll
      for (int i = 0; i < DH; ++i) s += q[i] * kr[i];
      s *= scale;
      const float nm   = fmaxf(m, s);
      const float corr = __expf(m - nm);
      const float e    = __expf(s - nm);
      ssum = ssum * corr + e;
      const float* vr = vbuf + j * DH;
#pragma unroll
      for (int i = 0; i < DH; ++i) o[i] = o[i] * corr + e * vr[i];
      m = nm;
    }
  }
  const float inv = 1.0f / ssum;
  float* op = Ob + ((size_t)(b * Tq + t)) * DMODEL + h * DH;
#pragma unroll
  for (int i = 0; i < DH; ++i) op[i] = o[i] * inv;
}

// =====================================================================
// Router: logits = x . router_w[e], top-2 softmax -> dense gate matrix [N,E]
// =====================================================================
__global__ __launch_bounds__(256)
void router_kernel(const float* __restrict__ Xn, const float* __restrict__ RW,
                   float* __restrict__ gates) {
  __shared__ float red[256];
  __shared__ float logits[NEXP];
  const int n = blockIdx.x, t = threadIdx.x;
  const int e = t >> 5, lane = t & 31;
  const float* xr = Xn + (size_t)n * DMODEL;
  const float* wr = RW + (size_t)e * DMODEL;
  float s = 0.0f;
  for (int d = lane; d < DMODEL; d += 32) s += xr[d] * wr[d];
  red[t] = s; __syncthreads();
  for (int s2 = 16; s2 > 0; s2 >>= 1) {
    if (lane < s2) red[t] += red[t + s2];
    __syncthreads();
  }
  if (lane == 0) logits[e] = red[t];
  __syncthreads();
  if (t == 0) {
    int i0 = 0; float v0 = logits[0];
    for (int i = 1; i < NEXP; ++i) if (logits[i] > v0) { v0 = logits[i]; i0 = i; }
    int i1 = -1; float v1 = -1e30f;
    for (int i = 0; i < NEXP; ++i)
      if (i != i0 && logits[i] > v1) { v1 = logits[i]; i1 = i; }
    const float e1 = __expf(v1 - v0);            // softmax over {v0, v1}
    const float inv = 1.0f / (1.0f + e1);
    for (int i = 0; i < NEXP; ++i) gates[(size_t)n * NEXP + i] = 0.0f;
    gates[(size_t)n * NEXP + i0] = inv;
    gates[(size_t)n * NEXP + i1] = e1 * inv;
  }
}

// =====================================================================
// G = silu(G) * U, elementwise
// =====================================================================
__global__ __launch_bounds__(256)
void silu_mul_kernel(float* __restrict__ G, const float* __restrict__ U, int nelem) {
  const int i = blockIdx.x * 256 + threadIdx.x;
  if (i < nelem) {
    const float g = G[i];
    G[i] = (g / (1.0f + __expf(-g))) * U[i];
  }
}

// =====================================================================
// Orchestration (graph-capture safe: only launches on `stream`)
// =====================================================================
extern "C" void kernel_launch(void* const* d_in, const int* in_sizes, int n_in,
                              void* d_out, int out_size, void* d_ws, size_t ws_size,
                              hipStream_t stream) {
  const int*   ids   = (const int*)  d_in[0];
  const float* enc   = (const float*)d_in[1];
  const float* emb   = (const float*)d_in[2];
  const float* n1w   = (const float*)d_in[3];
  const float* n2w   = (const float*)d_in[4];
  const float* n3w   = (const float*)d_in[5];
  const float* wqs   = (const float*)d_in[6];
  const float* wks   = (const float*)d_in[7];
  const float* wvs   = (const float*)d_in[8];
  const float* wos   = (const float*)d_in[9];
  const float* wqc   = (const float*)d_in[10];
  const float* wkc   = (const float*)d_in[11];
  const float* wvc   = (const float*)d_in[12];
  const float* woc   = (const float*)d_in[13];
  const float* rw    = (const float*)d_in[14];
  const float* wgate = (const float*)d_in[15];
  const float* wup   = (const float*)d_in[16];
  const float* wdown = (const float*)d_in[17];
  const float* fnw   = (const float*)d_in[18];
  const float* hw    = (const float*)d_in[19];
  float* out = (float*)d_out;

  // workspace carve-out (fp32 elements), ~67 MB total
  float* x     = (float*)d_ws;
  float* nb    = x  + (size_t)NTOK * DMODEL;
  float* qb    = nb + (size_t)NTOK * DMODEL;
  float* kb    = qb + (size_t)NTOK * DMODEL;
  float* vb    = kb + (size_t)NKV  * DMODEL;
  float* ab    = vb + (size_t)NKV  * DMODEL;
  float* gb    = ab + (size_t)NTOK * DMODEL;
  float* ub    = gb + (size_t)NTOK * HIDF;
  float* gates = ub + (size_t)NTOK * HIDF;

  const size_t DD = (size_t)DMODEL * DMODEL;
  const dim3 blk(256);
  const size_t attn_smem = 2u * 128u * DH * sizeof(float);   // 64 KB

#define GEMM(MODE, A_, W_, C_, N_, M_, K_, GATE_, GS_)                          \
  gemm_bf16_tn<MODE><<<dim3((M_) / 64, (N_) / 128), blk, 0, stream>>>(          \
      (A_), (W_), (C_), (N_), (M_), (K_), (GATE_), (GS_))

  embed_kernel<<<NTOK, blk, 0, stream>>>(ids, emb, x);

  for (int l = 0; l < NLAYER; ++l) {
    // ---- self attention: x += Attn(rmsnorm(x)) (causal) ----
    rmsnorm_kernel<<<NTOK, blk, 0, stream>>>(x, n1w + (size_t)l * DMODEL, nb);
    GEMM(0, nb, wqs + (size_t)l * DD, qb, NTOK, DMODEL, DMODEL, nullptr, 0);
    GEMM(0, nb, wks + (size_t)l * DD, kb, NTOK, DMODEL, DMODEL, nullptr, 0);
    GEMM(0, nb, wvs + (size_t)l * DD, vb, NTOK, DMODEL, DMODEL, nullptr, 0);
    attn_kernel<<<BB * NHEAD, TT, attn_smem, stream>>>(qb, kb, vb, ab, TT, TT, 1);
    GEMM(1, ab, wos + (size_t)l * DD, x, NTOK, DMODEL, DMODEL, nullptr, 0);

    // ---- cross attention: x += Attn(rmsnorm(x), enc_out) ----
    rmsnorm_kernel<<<NTOK, blk, 0, stream>>>(x, n2w + (size_t)l * DMODEL, nb);
    GEMM(0, nb,  wqc + (size_t)l * DD, qb, NTOK, DMODEL, DMODEL, nullptr, 0);
    GEMM(0, enc, wkc + (size_t)l * DD, kb, NKV,  DMODEL, DMODEL, nullptr, 0);
    GEMM(0, enc, wvc + (size_t)l * DD, vb, NKV,  DMODEL, DMODEL, nullptr, 0);
    attn_kernel<<<BB * NHEAD, TT, attn_smem, stream>>>(qb, kb, vb, ab, TT, TEE, 0);
    GEMM(1, ab, woc + (size_t)l * DD, x, NTOK, DMODEL, DMODEL, nullptr, 0);

    // ---- MoE: x += sum_e gate[n,e] * FFN_e(rmsnorm(x)), dense like reference ----
    rmsnorm_kernel<<<NTOK, blk, 0, stream>>>(x, n3w + (size_t)l * DMODEL, nb);
    router_kernel<<<NTOK, blk, 0, stream>>>(nb, rw + (size_t)l * NEXP * DMODEL, gates);
    for (int e = 0; e < NEXP; ++e) {
      const float* wg = wgate + ((size_t)l * NEXP + e) * HIDF * DMODEL;
      const float* wu = wup   + ((size_t)l * NEXP + e) * HIDF * DMODEL;
      const float* wd = wdown + ((size_t)l * NEXP + e) * DMODEL * HIDF;
      GEMM(0, nb, wg, gb, NTOK, HIDF, DMODEL, nullptr, 0);
      GEMM(0, nb, wu, ub, NTOK, HIDF, DMODEL, nullptr, 0);
      silu_mul_kernel<<<(NTOK * HIDF + 255) / 256, blk, 0, stream>>>(gb, ub, NTOK * HIDF);
      GEMM(2, gb, wd, x, NTOK, DMODEL, HIDF, gates + e, NEXP);
    }
  }

  // ---- final norm + 3 codebook heads, written straight into [L,B,T,K] ----
  rmsnorm_kernel<<<NTOK, blk, 0, stream>>>(x, fnw, nb);
  GEMM(3, nb, hw, out, NTOK, LLEV * KCB, DMODEL, nullptr, 0);
#undef GEMM
}
